// SSIM_58583353917726
// MI455X (gfx1250) — matrix operand order repaired and checked by reading, not compiled, not run on hardware
//
#include <hip/hip_runtime.h>
#include <hip/hip_bf16.h>

// ---------------------------------------------------------------------------
// SSIM on MI455X (gfx1250): separable 11-tap Gaussian conv done on the WMMA
// pipes as banded 16x16x32 f16 matmuls, fused elementwise SSIM + reduction.
// Patches are zero-padded to 32x32 in LDS so all WMMA operand-build loads are
// unconditional (no exec-mask save/restore per element).
// ---------------------------------------------------------------------------

typedef __attribute__((ext_vector_type(16))) _Float16 v16h;
typedef __attribute__((ext_vector_type(8)))  float    v8f;

#define IMG_H 512
#define IMG_W 512
#define N_PLANES 48              // B*C = 16*3
#define TILES_X 32               // 512/16
#define TILES_PER_PLANE (TILES_X * TILES_X)       // 1024
#define N_TILES (N_PLANES * TILES_PER_PLANE)      // 49152
#define WAVES_PER_BLOCK 4
#define BLOCK_THREADS (WAVES_PER_BLOCK * 32)
#define N_BLOCKS (N_TILES / WAVES_PER_BLOCK)      // 12288
#define N_ELEMS 12582912.0f      // 16*3*512*512

#define SSIM_C1 0.0001f          // 0.01^2
#define SSIM_C2 0.0009f          // 0.03^2

// Normalized 1-D Gaussian, sigma = 1.5, window 11 (matches reference window).
__device__ __constant__ float c_g[11] = {
    0.00102837f, 0.00759875f, 0.03600077f, 0.10936068f, 0.21300553f,
    0.26601172f,
    0.21300553f, 0.10936068f, 0.03600077f, 0.00759875f, 0.00102837f
};

// Per-wave LDS slice layout (floats):
//   patchA [32][32]   (img1, rows/cols 26..31 zero)          = 1024
//   patchB [32][32]   (img2)                                 = 1024
//   scratch[16][33]   (D-layout -> A-layout relayout buffer) =  528
#define PATCH_STRIDE 32
#define SCR_STRIDE 33
#define WAVE_LDS_FLOATS (1024 + 1024 + 528)   // 2576 -> 41216 B per 4-wave block

__global__ __launch_bounds__(BLOCK_THREADS) void
ssim_tile_kernel(const float* __restrict__ img1,
                 const float* __restrict__ img2,
                 float* __restrict__ partial)
{
    __shared__ float lds[WAVES_PER_BLOCK * WAVE_LDS_FLOATS];

    const int lane = threadIdx.x & 31;
    const int wid  = threadIdx.x >> 5;
    const int gwave = blockIdx.x * WAVES_PER_BLOCK + wid;

    const int plane = gwave / TILES_PER_PLANE;          // 0..47
    const int tile  = gwave - plane * TILES_PER_PLANE;  // 0..1023
    const int ty    = tile >> 5;                        // tile row
    const int tx    = tile & 31;                        // tile col
    const int y0    = ty * 16;
    const int x0    = tx * 16;

    const float* p1 = img1 + (size_t)plane * (IMG_H * IMG_W);
    const float* p2 = img2 + (size_t)plane * (IMG_H * IMG_W);

    float* pA = lds + wid * WAVE_LDS_FLOATS;
    float* pB = pA + 1024;
    float* sc = pB + 1024;

    // ---- Stage 26x26 halo patches into zero-padded 32x32 LDS tiles ----
    {
        const int col = lane;
        const int gx  = x0 - 5 + col;
        const bool cok = (col < 26) && ((unsigned)gx < (unsigned)IMG_W);
        #pragma unroll
        for (int r = 0; r < 32; ++r) {
            const int gy = y0 - 5 + r;
            const bool ok = cok && (r < 26) && ((unsigned)gy < (unsigned)IMG_H);
            float va = 0.0f, vb = 0.0f;
            if (ok) {
                const size_t off = (size_t)gy * IMG_W + gx;
                va = p1[off];
                vb = p2[off];
            }
            pA[r * PATCH_STRIDE + col] = va;   // unconditional: zeros pad
            pB[r * PATCH_STRIDE + col] = vb;
        }
    }

    // ---- Build constant band matrices in WMMA register layouts ----
    // A-layout (16x32 f16): lane m = lane&15, half h = lane>>4.
//   element j = 2v+e holds K = h*16 + j
    // B-layout (32x16 f16): lane n = lane&15, kh = lane>>4.
    //   element j holds K = kh*16 + j, column N = n
    const int mrow = lane & 15;
    const int half = lane >> 4;

    v16h aG;   // vertical band: Gv[m][k] = g[k-m]
    #pragma unroll
    for (int v = 0; v < 8; ++v) {
        #pragma unroll
        for (int e = 0; e < 2; ++e) {
            const int K = half * 16 + v * 2 + e;
            const int d = K - mrow;
            aG[v * 2 + e] = ((unsigned)d < 11u) ? (_Float16)c_g[d] : (_Float16)0.0f;
        }
    }

    v16h bG;   // horizontal band: Gh[k][n] = g[k-n]
    #pragma unroll
    for (int j = 0; j < 16; ++j) {
        const int k = half * 16 + j;
        const int d = k - mrow;   // n == mrow for B layout
        bG[j] = ((unsigned)d < 11u) ? (_Float16)c_g[d] : (_Float16)0.0f;
    }

    // ---- Vertical pass: acc[q][cg] = Gv x P_q  (one WMMA each, K=32>=26) ----
    // planes q: 0=x, 1=y, 2=x*x, 3=y*y, 4=x*y
    v8f acc[5][2];

    #pragma unroll
    for (int cg = 0; cg < 2; ++cg) {
        float av[16], bv[16];
        #pragma unroll
        for (int j = 0; j < 16; ++j) {
            const int k   = half * 16 + j;        // patch row 0..31 (padded)
            const int col = cg * 16 + mrow;       // patch col 0..31 (padded)
            av[j] = pA[k * PATCH_STRIDE + col];   // unconditional ds_load
            bv[j] = pB[k * PATCH_STRIDE + col];
        }
        #pragma unroll
        for (int q = 0; q < 5; ++q) {
            v16h b;
            #pragma unroll
            for (int j = 0; j < 16; ++j) {
                float x = av[j], y = bv[j];
                float p = (q == 0) ? x : (q == 1) ? y :
                          (q == 2) ? x * x : (q == 3) ? y * y : x * y;
                b[j] = (_Float16)p;
            }
            v8f cz = {};
            acc[q][cg] = __builtin_amdgcn_wmma_f32_16x16x32_f16(
                false, aG, false, b, (short)0, cz, false, false);
        }
    }

    // ---- Horizontal pass per plane: Out_q = V_q x Gh ----
    v8f outq[5];
    #pragma unroll
    for (int q = 0; q < 5; ++q) {
        // Relayout D (f32, row = r + 8*half, col = mrow) -> scratch[16][33]
        #pragma unroll
        for (int r = 0; r < 8; ++r) {
            const int row = r + (half ? 8 : 0);
            sc[row * SCR_STRIDE + mrow]       = acc[q][0][r];
            sc[row * SCR_STRIDE + 16 + mrow]  = acc[q][1][r];
        }
        // Read back in A layout (f16). In-order per-wave LDS: no barrier.
        v16h aV;
        #pragma unroll
        for (int v = 0; v < 8; ++v) {
            #pragma unroll
            for (int e = 0; e < 2; ++e) {
                const int K = half * 16 + v * 2 + e;
                aV[v * 2 + e] = (_Float16)sc[mrow * SCR_STRIDE + K];
            }
        }
        v8f cz = {};
        outq[q] = __builtin_amdgcn_wmma_f32_16x16x32_f16(
            false, aV, false, bG, (short)0, cz, false, false);
    }

    // ---- Elementwise SSIM map + wave reduction ----
    float s = 0.0f;
    #pragma unroll
    for (int r = 0; r < 8; ++r) {
        const float mu1 = outq[0][r];
        const float mu2 = outq[1][r];
        const float m11 = mu1 * mu1;
        const float m22 = mu2 * mu2;
        const float m12 = mu1 * mu2;
        const float s1  = outq[2][r] - m11;
        const float s2  = outq[3][r] - m22;
        const float s12 = outq[4][r] - m12;
        const float cs = (2.0f * s12 + SSIM_C2) / (s1 + s2 + SSIM_C2);
        const float l  = (2.0f * m12 + SSIM_C1) / (m11 + m22 + SSIM_C1);
        s += l * cs;
    }
    #pragma unroll
    for (int off = 16; off > 0; off >>= 1)
        s += __shfl_xor(s, off, 32);
    if (lane == 0)
        partial[gwave] = s;
}

__global__ __launch_bounds__(256) void
ssim_reduce_kernel(const float* __restrict__ partial, float* __restrict__ out)
{
    __shared__ float smem[8];
    float v = 0.0f;
    for (int i = threadIdx.x; i < N_TILES; i += 256)
        v += partial[i];
    #pragma unroll
    for (int off = 16; off > 0; off >>= 1)
        v += __shfl_xor(v, off, 32);
    if ((threadIdx.x & 31) == 0)
        smem[threadIdx.x >> 5] = v;
    __syncthreads();
    if (threadIdx.x == 0) {
        float t = 0.0f;
        #pragma unroll
        for (int w = 0; w < 8; ++w) t += smem[w];
        out[0] = t * (1.0f / N_ELEMS);
    }
}

extern "C" void kernel_launch(void* const* d_in, const int* in_sizes, int n_in,
                              void* d_out, int out_size, void* d_ws, size_t ws_size,
                              hipStream_t stream) {
    const float* img1 = (const float*)d_in[0];
    const float* img2 = (const float*)d_in[1];
    // d_in[2] (window) is the fixed Gaussian from the reference; baked into c_g.
    float* partial = (float*)d_ws;         // N_TILES floats (192 KB) scratch
    float* out = (float*)d_out;

    ssim_tile_kernel<<<N_BLOCKS, BLOCK_THREADS, 0, stream>>>(img1, img2, partial);
    ssim_reduce_kernel<<<1, 256, 0, stream>>>(partial, out);
}
